// SelfAttention_I_69355131896159
// MI455X (gfx1250) — matrix operand order repaired and checked by reading, not compiled, not run on hardware
//
#include <hip/hip_runtime.h>

typedef __attribute__((ext_vector_type(16))) _Float16 v16h;
typedef __attribute__((ext_vector_type(8)))  float    v8f;

#define HW   65536   // 256*256
#define BDIM 8

// ---------------------------------------------------------------------------
// block-wide sum reduction (256 threads)
// ---------------------------------------------------------------------------
__device__ inline float blockReduceSum(float v, float* s) {
  int t = threadIdx.x;
  s[t] = v; __syncthreads();
  for (int off = 128; off > 0; off >>= 1) {
    if (t < off) s[t] += s[t + off];
    __syncthreads();
  }
  float r = s[0];
  __syncthreads();
  return r;
}

// ---------------------------------------------------------------------------
// Pointwise conv = GEMM:  Y[b,co,p] = sum_ci W[co,ci] * X[b,ci,p] (+ bias)
// One wave computes a 16(pixel) x COUT tile via v_wmma_f32_16x16x32_f16.
// Weights staged in LDS as f16. 128 threads = 4 waves per block.
// ---------------------------------------------------------------------------
template<int COUT, int CIN>
__global__ void pwconv_k(const float* __restrict__ X, const float* __restrict__ Wm,
                         const float* __restrict__ bias, float* __restrict__ Y) {
  constexpr int MT = COUT / 16;
  constexpr int KT = CIN / 32;
  __shared__ _Float16 sW[COUT * CIN];
  __shared__ float    sB[COUT];
  int tid = threadIdx.x;
  for (int i = tid; i < COUT * CIN; i += 128) sW[i] = (_Float16)Wm[i];
  for (int i = tid; i < COUT;        i += 128) sB[i] = bias ? bias[i] : 0.0f;
  __syncthreads();

  int lane = tid & 31, wave = tid >> 5;
  bool hiL = lane >= 16;
  int  lr  = lane & 15;
  int  b   = blockIdx.y;
  int  p0  = (blockIdx.x * 4 + wave) * 16;

  v8f zero = {};
  v8f acc[MT];
#pragma unroll
  for (int m = 0; m < MT; ++m) acc[m] = zero;

#pragma unroll
  for (int kt = 0; kt < KT; ++kt) {
    // B fragment: X[k, p], lane<16 -> K=k0..k0+15, lane>=16 -> K=k0+16..k0+31
    v16h bv;
    int kb = kt * 32 + (hiL ? 16 : 0);
    const float* xp = X + (((size_t)b * CIN + kb) << 16) + p0 + lr;
#pragma unroll
    for (int r = 0; r < 16; ++r) bv[r] = (_Float16)xp[(size_t)r << 16];

#pragma unroll
    for (int m = 0; m < MT; ++m) {
      // A fragment from LDS: row M = m*16+lr ; K = base..base+15 (lane>=16: +16)
      v16h av;
      const _Float16* wp = sW + (m * 16 + lr) * CIN + kt * 32 + (hiL ? 16 : 0);
#pragma unroll
      for (int r = 0; r < 16; ++r) { av[r] = wp[r]; }
      acc[m] = __builtin_amdgcn_wmma_f32_16x16x32_f16(
          false, av, false, bv, (short)0, acc[m], false, false);
    }
  }

#pragma unroll
  for (int m = 0; m < MT; ++m) {
    int mrow = m * 16 + (hiL ? 8 : 0);
    float* yp = Y + (((size_t)b * COUT + mrow) << 16) + p0 + lr;
#pragma unroll
    for (int r = 0; r < 8; ++r) yp[(size_t)r << 16] = acc[m][r] + sB[mrow + r];
  }
}

// ---------------------------------------------------------------------------
// Depthwise conv (ks x ks, pad, dilation), H=W=256, one thread per output elem
// ---------------------------------------------------------------------------
__global__ void dwconv_k(const float* __restrict__ X, const float* __restrict__ Wt,
                         const float* __restrict__ bias, float* __restrict__ Y,
                         int C, int ks, int pad, int dil) {
  size_t idx = (size_t)blockIdx.x * 256 + threadIdx.x;
  size_t total = ((size_t)BDIM * C) << 16;
  if (idx >= total) return;
  int p  = (int)(idx & 65535);
  int c  = (int)((idx >> 16) % C);
  int b  = (int)((idx >> 16) / C);
  int hh = p >> 8, ww = p & 255;
  float acc = bias[c];
  const float* xb = X + (((size_t)b * C + c) << 16);
  const float* wc = Wt + c * ks * ks;
  for (int kh = 0; kh < ks; ++kh) {
    int ih = hh - pad + kh * dil;
    if ((unsigned)ih >= 256u) continue;
    for (int kw = 0; kw < ks; ++kw) {
      int iw = ww - pad + kw * dil;
      if ((unsigned)iw >= 256u) continue;
      acc += xb[(ih << 8) + iw] * wc[kh * ks + kw];
    }
  }
  Y[idx] = acc;
}

// ---------------------------------------------------------------------------
// GroupNorm (groups=2): two-stage deterministic reduction + fused affine+leaky
// ---------------------------------------------------------------------------
__global__ void gn_stage1(const float* __restrict__ X, float* __restrict__ partial,
                          int Cg) {
  __shared__ float sred[256];
  int bg = blockIdx.y, chunk = blockIdx.x;
  int b = bg >> 1, g = bg & 1;
  int C = Cg * 2;
  size_t base = (((size_t)b * C + (size_t)g * Cg) << 16);
  size_t L    = ((size_t)Cg) << 16;
  size_t span = L / gridDim.x;
  size_t s0   = (size_t)chunk * span;
  float s = 0.f, ss = 0.f;
  for (size_t i = threadIdx.x; i < span; i += 256) {
    float v = X[base + s0 + i];
    s += v; ss += v * v;
  }
  s  = blockReduceSum(s,  sred);
  ss = blockReduceSum(ss, sred);
  if (threadIdx.x == 0) {
    partial[((size_t)bg * gridDim.x + chunk) * 2]     = s;
    partial[((size_t)bg * gridDim.x + chunk) * 2 + 1] = ss;
  }
}

__global__ void gn_stage2(const float* __restrict__ partial, float* __restrict__ stats,
                          float L) {
  __shared__ float sred[256];
  int bg = blockIdx.x;
  float s  = partial[((size_t)bg * 256 + threadIdx.x) * 2];
  float ss = partial[((size_t)bg * 256 + threadIdx.x) * 2 + 1];
  s  = blockReduceSum(s,  sred);
  ss = blockReduceSum(ss, sred);
  if (threadIdx.x == 0) {
    float mean = s / L;
    float var  = ss / L - mean * mean;
    stats[bg * 2]     = mean;
    stats[bg * 2 + 1] = rsqrtf(var + 1e-5f);
  }
}

__global__ void gn_stage3(const float* __restrict__ X, const float* __restrict__ stats,
                          const float* __restrict__ gw, const float* __restrict__ gb,
                          float* __restrict__ Y, int Cg) {
  size_t idx = (size_t)blockIdx.x * 256 + threadIdx.x;
  int C = Cg * 2;
  int c = (int)((idx >> 16) % C);
  int b = (int)((idx >> 16) / C);
  int bg = b * 2 + (c >= Cg ? 1 : 0);
  float v = (X[idx] - stats[bg * 2]) * stats[bg * 2 + 1] * gw[c] + gb[c];
  Y[idx] = v >= 0.f ? v : 0.2f * v;   // LeakyReLU(0.2)
}

// ---------------------------------------------------------------------------
// L2 row norms over the pixel axis: 512 rows (b*64 + c), one block per row
// ---------------------------------------------------------------------------
__global__ void rownorm_k(const float* __restrict__ X, int Cs, int c0,
                          float* __restrict__ norms) {
  __shared__ float sred[256];
  int row = blockIdx.x;
  int b = row >> 6, c = row & 63;
  const float* xr = X + (((size_t)b * Cs + c0 + c) << 16);
  float ss = 0.f;
  for (int i = threadIdx.x; i < HW; i += 256) { float v = xr[i]; ss += v * v; }
  ss = blockReduceSum(ss, sred);
  if (threadIdx.x == 0) norms[row] = sqrtf(ss);
}

// ---------------------------------------------------------------------------
// Partial Gram: G[bh,i,j] = sum_n q[i,n]*k[j,n] via WMMA over K=65536.
// K split over gridDim.y chunks (16) for occupancy/bandwidth: 512 workgroups.
// Block = 8 waves; each wave accumulates a partial 16x16, LDS-combined, then
// written to pg[(chunk*32 + bh)*256 + i*16 + j]. Deterministic (no atomics).
// ---------------------------------------------------------------------------
__global__ void gram_partial_k(const float* __restrict__ q, const float* __restrict__ k,
                               float* __restrict__ pg) {
  __shared__ float sred[8][32][8];
  int tid = threadIdx.x;
  int lane = tid & 31, wave = tid >> 5;
  int bh = blockIdx.x, b = bh >> 2, h = bh & 3;
  int chunk = blockIdx.y;                    // 0..15
  bool hiL = lane >= 16;
  int  lr  = lane & 15;
  const float* qrow = q + (((size_t)b * 64  + h * 16 + lr) << 16);  // A: M row
  const float* krow = k + (((size_t)b * 128 + h * 16 + lr) << 16);  // B: N col

  const int KSTEPS = HW / 32;                // 2048 k-steps of 32 pixels
  const int PERCHUNK = KSTEPS / 16;          // 128 per chunk
  int kt0 = chunk * PERCHUNK;

  v8f acc = {};
  for (int kt = kt0 + wave; kt < kt0 + PERCHUNK; kt += 8) {
    int p0 = kt << 5;
    v16h av, bv;
    int ao = p0 + (hiL ? 16 : 0);
#pragma unroll
    for (int r = 0; r < 16; ++r) {
      av[r]     = (_Float16)qrow[ao + r];
    }
    int bo = p0 + (hiL ? 16 : 0);
#pragma unroll
    for (int r = 0; r < 16; ++r) bv[r] = (_Float16)krow[bo + r];
    acc = __builtin_amdgcn_wmma_f32_16x16x32_f16(
        false, av, false, bv, (short)0, acc, false, false);
  }
#pragma unroll
  for (int r = 0; r < 8; ++r) sred[wave][lane][r] = acc[r];
  __syncthreads();
  if (wave == 0) {
#pragma unroll
    for (int r = 0; r < 8; ++r) {
      float s = 0.f;
      for (int w2 = 0; w2 < 8; ++w2) s += sred[w2][lane][r];
      int i = (hiL ? 8 : 0) + r;     // D layout: VGPR r -> M
      int j = lr;                    // lane -> N
      pg[((size_t)chunk * 32 + bh) * 256 + i * 16 + j] = s;
    }
  }
}

// Fold 16 partial Grams: one block per bh, one thread per (i,j)
__global__ void gram_reduce_k(const float* __restrict__ pg, float* __restrict__ gram) {
  int bh = blockIdx.x;
  int e  = threadIdx.x;   // 0..255 -> (i,j)
  float s = 0.f;
#pragma unroll
  for (int c = 0; c < 16; ++c) s += pg[((size_t)c * 32 + bh) * 256 + e];
  gram[(size_t)bh * 256 + e] = s;
}

// ---------------------------------------------------------------------------
// attn = softmax( G / (|q_i||k_j|) * temperature[h] )  — 16 rows per (b,h)
// ---------------------------------------------------------------------------
__global__ void attn_softmax_k(const float* __restrict__ gram,
                               const float* __restrict__ nq,
                               const float* __restrict__ nk,
                               const float* __restrict__ temp,
                               float* __restrict__ attn) {
  int bh = blockIdx.x, b = bh >> 2, h = bh & 3;
  int i = threadIdx.x;  // 16 threads
  float t  = temp[h];
  float qn = fmaxf(nq[b * 64 + h * 16 + i], 1e-12f);
  float row[16];
  float m = -1e30f;
  for (int j = 0; j < 16; ++j) {
    float kn = fmaxf(nk[b * 64 + h * 16 + j], 1e-12f);
    float a = gram[((size_t)bh * 16 + i) * 16 + j] / (qn * kn) * t;
    row[j] = a; m = fmaxf(m, a);
  }
  float s = 0.f;
  for (int j = 0; j < 16; ++j) { row[j] = __expf(row[j] - m); s += row[j]; }
  float inv = 1.f / s;
  for (int j = 0; j < 16; ++j) attn[((size_t)bh * 16 + i) * 16 + j] = row[j] * inv;
}

// ---------------------------------------------------------------------------
// out[b, h*16+i, p] = sum_j attn[b,h,i,j] * v[b,h,j,p]  (v at channel 64 of kv)
// ---------------------------------------------------------------------------
__global__ void av_k(const float* __restrict__ attn, const float* __restrict__ kv,
                     float* __restrict__ out) {
  __shared__ float sA[4 * 16 * 16];
  int b = blockIdx.y;
  for (int i = threadIdx.x; i < 1024; i += 256) sA[i] = attn[(size_t)b * 1024 + i];
  __syncthreads();
  int p = blockIdx.x * 256 + threadIdx.x;
  for (int h = 0; h < 4; ++h) {
    float v[16];
#pragma unroll
    for (int j = 0; j < 16; ++j)
      v[j] = kv[(((size_t)b * 128 + 64 + h * 16 + j) << 16) + p];
#pragma unroll
    for (int i2 = 0; i2 < 16; ++i2) {
      float a = 0.f;
#pragma unroll
      for (int j = 0; j < 16; ++j) a += sA[(h * 16 + i2) * 16 + j] * v[j];
      out[(((size_t)b * 64 + h * 16 + i2) << 16) + p] = a;
    }
  }
}

// ---------------------------------------------------------------------------
extern "C" void kernel_launch(void* const* d_in, const int* in_sizes, int n_in,
                              void* d_out, int out_size, void* d_ws, size_t ws_size,
                              hipStream_t stream) {
  (void)in_sizes; (void)n_in; (void)out_size; (void)ws_size;
  const float* x      = (const float*)d_in[0];
  const float* temp   = (const float*)d_in[1];
  const float* w_kv   = (const float*)d_in[2];
  const float* kv_c0w = (const float*)d_in[3];
  const float* kv_c0b = (const float*)d_in[4];
  const float* kv_csw = (const float*)d_in[5];
  const float* kv_csb = (const float*)d_in[6];
  const float* kv_c1w = (const float*)d_in[7];
  const float* kv_c1b = (const float*)d_in[8];
  const float* kv_gnw = (const float*)d_in[9];
  const float* kv_gnb = (const float*)d_in[10];
  const float* w_q    = (const float*)d_in[11];
  const float* q_c0w  = (const float*)d_in[12];
  const float* q_c0b  = (const float*)d_in[13];
  const float* q_csw  = (const float*)d_in[14];
  const float* q_csb  = (const float*)d_in[15];
  const float* q_c1w  = (const float*)d_in[16];
  const float* q_c1b  = (const float*)d_in[17];
  const float* q_gnw  = (const float*)d_in[18];
  const float* q_gnb  = (const float*)d_in[19];
  const float* w_proj = (const float*)d_in[20];
  float* out = (float*)d_out;

  const size_t P128 = (size_t)BDIM * 128 * HW;   // 67,108,864 floats
  const size_t P64  = P128 / 2;
  float* buf1    = (float*)d_ws;        // 268 MB
  float* buf2    = buf1 + P128;         // 268 MB
  float* partial = buf2 + P128;         // 16*256*2
  float* stats   = partial + 8192;      // 32
  float* normq   = stats + 32;          // 512
  float* normk   = normq + 512;         // 512
  float* gram    = normk + 512;         // 32*16*16
  float* attn    = gram + 8192;         // 32*16*16
  float* pgram   = attn + 8192;         // 16*32*256 = 131072 (512 KB)

  dim3 gGemm(1024, BDIM);

  // ---- kv branch: conv1x1 -> dw5x5 -> dw3x3(d3) -> conv1x1 -> GN+leaky ----
  pwconv_k<128, 64><<<gGemm, 128, 0, stream>>>(x, w_kv, nullptr, buf1);
  dwconv_k<<<(unsigned)(P128 / 256), 256, 0, stream>>>(buf1, kv_c0w, kv_c0b, buf2, 128, 5, 2, 1);
  dwconv_k<<<(unsigned)(P128 / 256), 256, 0, stream>>>(buf2, kv_csw, kv_csb, buf1, 128, 3, 3, 3);
  pwconv_k<128, 128><<<gGemm, 128, 0, stream>>>(buf1, kv_c1w, kv_c1b, buf2);
  gn_stage1<<<dim3(256, 16), 256, 0, stream>>>(buf2, partial, 64);
  gn_stage2<<<16, 256, 0, stream>>>(partial, stats, (float)((size_t)64 << 16));
  gn_stage3<<<(unsigned)(P128 / 256), 256, 0, stream>>>(buf2, stats, kv_gnw, kv_gnb, buf1, 64);
  // buf1 now holds kv (k = channels 0..63, v = channels 64..127)

  // ---- q branch (ping-pongs inside the freed buf2) ----
  float* qA = buf2;
  float* qB = buf2 + P64;
  pwconv_k<64, 64><<<gGemm, 128, 0, stream>>>(x, w_q, nullptr, qA);
  dwconv_k<<<(unsigned)(P64 / 256), 256, 0, stream>>>(qA, q_c0w, q_c0b, qB, 64, 5, 2, 1);
  dwconv_k<<<(unsigned)(P64 / 256), 256, 0, stream>>>(qB, q_csw, q_csb, qA, 64, 3, 3, 3);
  pwconv_k<64, 64><<<gGemm, 128, 0, stream>>>(qA, q_c1w, q_c1b, qB);
  gn_stage1<<<dim3(256, 16), 256, 0, stream>>>(qB, partial, 32);
  gn_stage2<<<16, 256, 0, stream>>>(partial, stats, (float)((size_t)32 << 16));
  gn_stage3<<<(unsigned)(P64 / 256), 256, 0, stream>>>(qB, stats, q_gnw, q_gnb, qA, 32);
  // qA now holds q

  // ---- channel attention ----
  rownorm_k<<<512, 256, 0, stream>>>(qA,   64, 0, normq);
  rownorm_k<<<512, 256, 0, stream>>>(buf1, 128, 0, normk);
  gram_partial_k<<<dim3(32, 16), 256, 0, stream>>>(qA, buf1, pgram);
  gram_reduce_k<<<32, 256, 0, stream>>>(pgram, gram);
  attn_softmax_k<<<32, 16, 0, stream>>>(gram, normq, normk, temp, attn);
  av_k<<<dim3(256, BDIM), 256, 0, stream>>>(attn, buf1, qB);

  // ---- final 1x1 projection into d_out ----
  pwconv_k<64, 64><<<gGemm, 128, 0, stream>>>(qB, w_proj, nullptr, out);
}
